// ARC_70583492542658
// MI455X (gfx1250) — matrix operand order repaired and checked by reading, not compiled, not run on hardware
//
#include <hip/hip_runtime.h>
#include <hip/hip_bf16.h>
#include <stdint.h>

// ---- problem constants (match reference) ----
#define Bn 8
#define Sn 2048
#define SLn 128
#define DIn 1024
#define DKn 64
#define DVn 64
#define Ln (Sn + 2 * SLn)        // 2304
#define NEGF (-1.0e30f)
#define SCALEF 0.125f            // DK^-0.5
#define KT 32                    // keys per streamed tile

typedef __bf16 bf16;
typedef __attribute__((ext_vector_type(16))) __bf16 v16bf;
typedef __attribute__((ext_vector_type(8)))  __bf16 v8bf;
typedef __attribute__((ext_vector_type(8)))  float  v8f;

// ---------------- WMMA helper (CDNA5 16x16x32 bf16, f32 accum) ----------------
__device__ __forceinline__ v8f wmma_bf16(v16bf a, v16bf b, v8f c) {
  return __builtin_amdgcn_wmma_f32_16x16x32_bf16(false, a, false, b, (short)0, c, false, false);
}

// A-fragment (16x32 bf16, row-major source, stride lda elements), this code's contiguous k order:
// lane r=lane&15 -> row M=r ; hi=lane>>4 ; a[e]:K=16*hi+e
__device__ __forceinline__ v16bf load_a_frag(const bf16* A, int lda, int k0) {
  int lane = threadIdx.x & 31;
  int r = lane & 15, hi = lane >> 4;
  const v8bf x0 = *(const v8bf*)(A + (size_t)r * lda + k0 + 16 * hi);
  const v8bf x1 = *(const v8bf*)(A + (size_t)r * lda + k0 + 16 * hi + 8);
  v16bf a;
#pragma unroll
  for (int e = 0; e < 8; ++e) { a[e] = x0[e]; a[8 + e] = x1[e]; }
  return a;
}

// B-fragment (32x16) from generic row-major W (K x N, stride ldn), strided reads
__device__ __forceinline__ v16bf load_b_frag(const bf16* W, int ldn, int k0, int n0) {
  int lane = threadIdx.x & 31;
  int n = lane & 15, hi = lane >> 4;
  const bf16* p = W + (size_t)(k0 + 16 * hi) * ldn + n0 + n;
  v16bf b;
#pragma unroll
  for (int e = 0; e < 16; ++e) b[e] = p[(size_t)e * ldn];
  return b;
}

// B-fragment from pre-swizzled (fragment-major) weights: one contiguous 32B load per lane
__device__ __forceinline__ v16bf load_b_frag_sw(const bf16* Wsw, int N, int k0, int n0) {
  int lane = threadIdx.x & 31;
  int frag = (k0 >> 5) * (N >> 4) + (n0 >> 4);
  return *(const v16bf*)(Wsw + (size_t)frag * 512 + lane * 16);
}

// B-fragment of K^T (dk x keys) from row-major K (keys x DK): contiguous 16 bf16 per lane
__device__ __forceinline__ v16bf load_bT_frag(const bf16* K, int ldk, int keybase, int dk0) {
  int lane = threadIdx.x & 31;
  int n = lane & 15, hi = lane >> 4;
  return *(const v16bf*)(K + (size_t)(keybase + n) * ldk + dk0 + 16 * hi);
}

// ---- CDNA5 async global->LDS copy (16B per lane) + counter waits ----
__device__ __forceinline__ void async_load_b128(uint32_t lds_addr, uint64_t gaddr) {
  asm volatile("global_load_async_to_lds_b128 %0, %1, off"
               :: "v"(lds_addr), "v"(gaddr) : "memory");
}
__device__ __forceinline__ void wait_async0() {
  asm volatile("s_wait_asynccnt 0x0" ::: "memory");
}
__device__ __forceinline__ void wait_ds0() {
  asm volatile("s_wait_dscnt 0x0" ::: "memory");
}

// ---------------- 1) fp32 -> bf16 convert (plain) ----------------
__global__ void cvt_bf16_kernel(const float* __restrict__ src, bf16* __restrict__ dst, int n) {
  int i = blockIdx.x * blockDim.x + threadIdx.x;
  if (i < n) dst[i] = (bf16)src[i];
}

// ---------------- 1b) fp32 -> bf16 convert with B-fragment swizzle ----------------
// dst[frag*512 + lane*16 + e] = src[k][n], frag=(k/32)*(N/16)+(n/16), lane=((k>>4)&1)*16+(n&15), e=k&15
__global__ void cvt_swz_kernel(const float* __restrict__ src, bf16* __restrict__ dst, int K, int N) {
  int i = blockIdx.x * blockDim.x + threadIdx.x;
  if (i >= K * N) return;
  int k = i / N, n = i % N;
  int frag = (k >> 5) * (N >> 4) + (n >> 4);
  int lane = (((k >> 4) & 1) << 4) + (n & 15);
  int e = k & 15;
  dst[(size_t)frag * 512 + lane * 16 + e] = (bf16)src[i];
}

// ---------------- 2) segment LayerNorm -> bf16 ----------------
__global__ void ln_kernel(const float* __restrict__ x,
                          const float* g_in, const float* b_in,
                          const float* g_ss, const float* b_ss,
                          const float* g_se, const float* b_se,
                          bf16* __restrict__ xn) {
  __shared__ float rs[8], rs2[8];
  int row = blockIdx.x;              // 0 .. B*L-1
  int t = row % Ln;
  const float* g; const float* bb;
  if (t < SLn)            { g = g_ss; bb = b_ss; }
  else if (t >= Ln - SLn) { g = g_se; bb = b_se; }
  else                    { g = g_in; bb = b_in; }
  const float* xr = x + (size_t)row * DIn;
  float s = 0.f, s2 = 0.f;
  for (int i = threadIdx.x; i < DIn; i += 256) { float v = xr[i]; s += v; s2 += v * v; }
#pragma unroll
  for (int d = 1; d < 32; d <<= 1) { s += __shfl_xor(s, d, 32); s2 += __shfl_xor(s2, d, 32); }
  if ((threadIdx.x & 31) == 0) { rs[threadIdx.x >> 5] = s; rs2[threadIdx.x >> 5] = s2; }
  __syncthreads();
  float ts = 0.f, ts2 = 0.f;
#pragma unroll
  for (int w = 0; w < 8; ++w) { ts += rs[w]; ts2 += rs2[w]; }
  float mean = ts * (1.0f / DIn);
  float var  = ts2 * (1.0f / DIn) - mean * mean;
  float rstd = rsqrtf(var + 1e-5f);
  bf16* xo = xn + (size_t)row * DIn;
  for (int i = threadIdx.x; i < DIn; i += 256)
    xo[i] = (bf16)((xr[i] - mean) * rstd * g[i] + bb[i]);
}

// ---------------- 3) QKV projection GEMM (xn @ W_seg), swizzled weights ----------------
__global__ __launch_bounds__(256) void qkv_kernel(const bf16* __restrict__ xn,
                                                  const bf16* __restrict__ Wcat,  // swizzled [seg][qkv]
                                                  bf16* __restrict__ Q, bf16* __restrict__ K,
                                                  bf16* __restrict__ V) {
  int wave = threadIdx.x >> 5, lane = threadIdx.x & 31;
  int rowtile = blockIdx.x * 8 + wave;
  int mat = blockIdx.y;
  int row = rowtile * 16;                 // global row in [0, B*L)
  int t = row % Ln;
  int seg = (t < SLn) ? 1 : ((t >= Ln - SLn) ? 2 : 0);
  const bf16* W = Wcat + ((size_t)seg * 3 + mat) * (DIn * DKn);
  const bf16* A = xn + (size_t)row * DIn;
  v8f c0{}, c1{}, c2{}, c3{};
  for (int k0 = 0; k0 < DIn; k0 += 32) {
    v16bf a = load_a_frag(A, DIn, k0);
    v16bf bb;
    bb = load_b_frag_sw(W, DKn, k0, 0);  c0 = wmma_bf16(a, bb, c0);
    bb = load_b_frag_sw(W, DKn, k0, 16); c1 = wmma_bf16(a, bb, c1);
    bb = load_b_frag_sw(W, DKn, k0, 32); c2 = wmma_bf16(a, bb, c2);
    bb = load_b_frag_sw(W, DKn, k0, 48); c3 = wmma_bf16(a, bb, c3);
  }
  bf16* out = (mat == 0 ? Q : (mat == 1 ? K : V)) + (size_t)row * DKn;
  int hi = lane >> 4, nn = lane & 15;
#pragma unroll
  for (int i = 0; i < 8; ++i) {
    int M = i + 8 * hi;
    out[M * DKn + nn]      = (bf16)c0[i];
    out[M * DKn + 16 + nn] = (bf16)c1[i];
    out[M * DKn + 32 + nn] = (bf16)c2[i];
    out[M * DKn + 48 + nn] = (bf16)c3[i];
  }
}

// ---------------- 4) gate row totals T[q] = sum_j sigmoid(q . k_j) (middle) ----------------
__global__ __launch_bounds__(256) void gate_total_kernel(const bf16* __restrict__ Q,
                                                         const bf16* __restrict__ K,
                                                         float* __restrict__ Tg) {
  int wave = threadIdx.x >> 5, lane = threadIdx.x & 31;
  int b = blockIdx.x;
  int rt = blockIdx.y * 8 + wave;          // 0..127
  int q0 = SLn + rt * 16;                  // global middle row
  const bf16* Qb = Q + (size_t)b * Ln * DKn;
  const bf16* Kb = K + (size_t)b * Ln * DKn;
  v16bf a0 = load_a_frag(Qb + (size_t)q0 * DKn, DKn, 0);
  v16bf a1 = load_a_frag(Qb + (size_t)q0 * DKn, DKn, 32);
  int hi = lane >> 4, nn = lane & 15;
  float acc[8];
#pragma unroll
  for (int i = 0; i < 8; ++i) acc[i] = 0.f;
  for (int k0 = SLn; k0 < Ln - SLn; k0 += 32) {
    v8f s0{}, s1{};
    v16bf bb;
    bb = load_bT_frag(Kb, DKn, k0, 0);       s0 = wmma_bf16(a0, bb, s0);
    bb = load_bT_frag(Kb, DKn, k0, 32);      s0 = wmma_bf16(a1, bb, s0);
    bb = load_bT_frag(Kb, DKn, k0 + 16, 0);  s1 = wmma_bf16(a0, bb, s1);
    bb = load_bT_frag(Kb, DKn, k0 + 16, 32); s1 = wmma_bf16(a1, bb, s1);
#pragma unroll
    for (int i = 0; i < 8; ++i)
      acc[i] += 1.f / (1.f + __expf(-s0[i])) + 1.f / (1.f + __expf(-s1[i]));
  }
#pragma unroll
  for (int i = 0; i < 8; ++i) {
    float v = acc[i];
#pragma unroll
    for (int d = 1; d < 16; d <<= 1) v += __shfl_xor(v, d, 16);
    if (nn == 0) Tg[(size_t)b * Sn + (q0 - SLn) + i + 8 * hi] = v;
  }
}

// ---------------- 5) li = q_mid @ cope_emb  -> (B,S,SL) fp32 (swizzled emb) ----------------
__global__ __launch_bounds__(256) void li_kernel(const bf16* __restrict__ Q,
                                                 const bf16* __restrict__ emb,  // swizzled DKxSL
                                                 float* __restrict__ Li) {
  int wave = threadIdx.x >> 5, lane = threadIdx.x & 31;
  int b = blockIdx.x;
  int rt = blockIdx.y * 8 + wave;          // 0..127
  int q0 = SLn + rt * 16;
  const bf16* Qb = Q + (size_t)b * Ln * DKn;
  v16bf a0 = load_a_frag(Qb + (size_t)q0 * DKn, DKn, 0);
  v16bf a1 = load_a_frag(Qb + (size_t)q0 * DKn, DKn, 32);
  int hi = lane >> 4, nn = lane & 15;
  float* Lb = Li + ((size_t)b * Sn + (q0 - SLn)) * SLn;
#pragma unroll
  for (int j = 0; j < 8; ++j) {
    v8f c{};
    v16bf bb;
    bb = load_b_frag_sw(emb, SLn, 0, j * 16);  c = wmma_bf16(a0, bb, c);
    bb = load_b_frag_sw(emb, SLn, 32, j * 16); c = wmma_bf16(a1, bb, c);
#pragma unroll
    for (int i = 0; i < 8; ++i) Lb[(size_t)(i + 8 * hi) * SLn + j * 16 + nn] = c[i];
  }
}

// ---------------- 6) RoPE (pairs d, d+32) ----------------
__global__ void rope_kernel(const bf16* __restrict__ Q, const bf16* __restrict__ K,
                            const int* __restrict__ offset,
                            bf16* __restrict__ Qr, bf16* __restrict__ Kr) {
  int idx = blockIdx.x * blockDim.x + threadIdx.x;   // B*L*32
  if (idx >= Bn * Ln * 32) return;
  int d = idx & 31;
  int bt = idx >> 5;
  int t = bt % Ln;
  float inv = __powf(10000.f, -((float)(2 * d)) / (float)DKn);
  float ang = (float)(t + offset[0]) * inv;
  float c = __cosf(ang), s = __sinf(ang);
  size_t base = (size_t)bt * DKn;
  float q1 = (float)Q[base + d], q2 = (float)Q[base + d + 32];
  Qr[base + d]      = (bf16)(q1 * c - q2 * s);
  Qr[base + d + 32] = (bf16)(q1 * s + q2 * c);
  float k1 = (float)K[base + d], k2 = (float)K[base + d + 32];
  Kr[base + d]      = (bf16)(k1 * c - k2 * s);
  Kr[base + d + 32] = (bf16)(k1 * s + k2 * c);
}

// ---------------- 7) fused flash attention with CoPE bias ----------------
// grid = (B, 18), 256 threads = 8 waves. Block q-range = 128 rows; keys streamed through LDS
// via async global->LDS copies, double-buffered; all 8 waves share each K/V/Kn tile.
__global__ __launch_bounds__(256) void attn_kernel(const bf16* __restrict__ Qr,
                                                   const bf16* __restrict__ Kr,
                                                   const bf16* __restrict__ V,
                                                   const bf16* __restrict__ Qn,   // pre-RoPE
                                                   const bf16* __restrict__ Kn,   // pre-RoPE
                                                   const float* __restrict__ Tg,
                                                   const float* __restrict__ Li,
                                                   float* __restrict__ out) {
  __shared__ __align__(16) bf16 sK [2][KT * DKn];   // 2 x 4KB (RoPE'd K)
  __shared__ __align__(16) bf16 sV [2][KT * DKn];   // 2 x 4KB
  __shared__ __align__(16) bf16 sKn[2][KT * DKn];   // 2 x 4KB (pre-RoPE K for gates)
  __shared__ __align__(16) bf16 pstage[8][16 * 32]; // 8KB
  int tid = threadIdx.x;
  int wave = tid >> 5, lane = tid & 31;
  int b = blockIdx.x;
  int qblk = blockIdx.y * 128;            // block q range [qblk, qblk+128)
  int qbase = qblk + wave * 16;
  const bf16* Qb  = Qr + (size_t)b * Ln * DKn;
  const bf16* Kb  = Kr + (size_t)b * Ln * DKn;
  const bf16* Vb  = V  + (size_t)b * Ln * DKn;
  const bf16* Qnb = Qn + (size_t)b * Ln * DKn;
  const bf16* Knb = Kn + (size_t)b * Ln * DKn;
  int hi = lane >> 4, nn = lane & 15;

  v16bf aq0 = load_a_frag(Qb + (size_t)qbase * DKn, DKn, 0);
  v16bf aq1 = load_a_frag(Qb + (size_t)qbase * DKn, DKn, 32);
  bool qmid = (qbase >= SLn) && (qbase < Ln - SLn);
  v16bf an0{}, an1{};
  if (qmid) {
    an0 = load_a_frag(Qnb + (size_t)qbase * DKn, DKn, 0);
    an1 = load_a_frag(Qnb + (size_t)qbase * DKn, DKn, 32);
  }

  float m_i[8], l_i[8], P_i[8], T_i[8];
  v8f O0{}, O1{}, O2{}, O3{};
#pragma unroll
  for (int i = 0; i < 8; ++i) {
    m_i[i] = -3.0e38f; l_i[i] = 0.f; P_i[i] = 0.f;
    int q = qbase + i + 8 * hi;
    T_i[i] = qmid ? Tg[(size_t)b * Sn + (q - SLn)] : 0.f;
  }
  const float* Lib = Li + (size_t)b * Sn * SLn;

  // LDS byte offsets for async copy destinations (this thread's 16B slice)
  uint32_t oK [2] = { (uint32_t)(uintptr_t)&sK [0][0] + (uint32_t)tid * 16,
                      (uint32_t)(uintptr_t)&sK [1][0] + (uint32_t)tid * 16 };
  uint32_t oV [2] = { (uint32_t)(uintptr_t)&sV [0][0] + (uint32_t)tid * 16,
                      (uint32_t)(uintptr_t)&sV [1][0] + (uint32_t)tid * 16 };
  uint32_t oKn[2] = { (uint32_t)(uintptr_t)&sKn[0][0] + (uint32_t)tid * 16,
                      (uint32_t)(uintptr_t)&sKn[1][0] + (uint32_t)tid * 16 };

  int nkt = qblk / KT + 4;                // covers keys up to qblk+127
  // prologue: issue tile 0 into buffer 0 (256 threads x 16B = 4KB per tensor)
  {
    uint64_t goff = (uint64_t)tid * 16;
    async_load_b128(oK [0], (uint64_t)(uintptr_t)Kb  + goff);
    async_load_b128(oV [0], (uint64_t)(uintptr_t)Vb  + goff);
    async_load_b128(oKn[0], (uint64_t)(uintptr_t)Knb + goff);
  }

  for (int kt = 0; kt < nkt; ++kt) {
    int cur = kt & 1;
    int k0 = kt * KT;
    wait_async0();                        // this wave's slice of tile kt is in LDS
    __syncthreads();                      // all waves' slices are in LDS
    if (kt + 1 < nkt) {                   // overlap copy of tile kt+1 with compute of kt
      int nb = cur ^ 1;
      uint64_t goff = (uint64_t)(kt + 1) * (KT * DKn * 2) + (uint64_t)tid * 16;
      async_load_b128(oK [nb], (uint64_t)(uintptr_t)Kb  + goff);
      async_load_b128(oV [nb], (uint64_t)(uintptr_t)Vb  + goff);
      async_load_b128(oKn[nb], (uint64_t)(uintptr_t)Knb + goff);
    }
    if (qbase + 15 >= k0) {               // this wave still has causal work in this tile
      const bf16* Kt  = &sK [cur][0];
      const bf16* Vt  = &sV [cur][0];
      const bf16* Knt = &sKn[cur][0];
      // ---- scores: 16x32 via 4 WMMA (K^T fragments from LDS) ----
      v8f s0{}, s1{};
      {
        v16bf bk;
        bk = load_bT_frag(Kt, DKn, 0, 0);    s0 = wmma_bf16(aq0, bk, s0);
        bk = load_bT_frag(Kt, DKn, 0, 32);   s0 = wmma_bf16(aq1, bk, s0);
        bk = load_bT_frag(Kt, DKn, 16, 0);   s1 = wmma_bf16(aq0, bk, s1);
        bk = load_bT_frag(Kt, DKn, 16, 32);  s1 = wmma_bf16(aq1, bk, s1);
      }
      bool kmid = (k0 >= SLn) && (k0 < Ln - SLn);
      bool dogate = qmid && kmid;
      v8f g0{}, g1{};
      if (dogate) {
        v16bf bk;
        bk = load_bT_frag(Knt, DKn, 0, 0);   g0 = wmma_bf16(an0, bk, g0);
        bk = load_bT_frag(Knt, DKn, 0, 32);  g0 = wmma_bf16(an1, bk, g0);
        bk = load_bT_frag(Knt, DKn, 16, 0);  g1 = wmma_bf16(an0, bk, g1);
        bk = load_bT_frag(Knt, DKn, 16, 32); g1 = wmma_bf16(an1, bk, g1);
      }
#pragma unroll
      for (int i = 0; i < 8; ++i) {
        int q = qbase + i + 8 * hi;
        float sc0 = s0[i] * SCALEF, sc1 = s1[i] * SCALEF;
        if (dogate) {
          float gg0 = 1.f / (1.f + __expf(-g0[i]));
          float gg1 = 1.f / (1.f + __expf(-g1[i]));
          // exclusive prefix across this row's 32 columns (16-lane scans)
          float inc0 = gg0;
#pragma unroll
          for (int d = 1; d < 16; d <<= 1) { float tv = __shfl_up(inc0, d, 16); if (nn >= d) inc0 += tv; }
          float tot0 = __shfl(inc0, 15, 16);
          float ex0 = inc0 - gg0;
          float inc1 = gg1;
#pragma unroll
          for (int d = 1; d < 16; d <<= 1) { float tv = __shfl_up(inc1, d, 16); if (nn >= d) inc1 += tv; }
          float tot1 = __shfl(inc1, 15, 16);
          float ex1 = inc1 - gg1 + tot0;
          float pos0 = fmaxf(fminf(T_i[i] - P_i[i] - ex0, (float)(SLn - 1)), 0.f);
          float pos1 = fmaxf(fminf(T_i[i] - P_i[i] - ex1, (float)(SLn - 1)), 0.f);
          const float* lr = Lib + (size_t)(q - SLn) * SLn;
          { float pf = floorf(pos0); float w = pos0 - pf;
            sc0 += lr[(int)ceilf(pos0)] * w + lr[(int)pf] * (1.f - w); }
          { float pf = floorf(pos1); float w = pos1 - pf;
            sc1 += lr[(int)ceilf(pos1)] * w + lr[(int)pf] * (1.f - w); }
          P_i[i] += tot0 + tot1;
        }
        // causal mask
        if (k0 + nn > q)      sc0 = NEGF;
        if (k0 + 16 + nn > q) sc1 = NEGF;
        // online softmax row update
        float rm = fmaxf(sc0, sc1);
#pragma unroll
        for (int d = 1; d < 16; d <<= 1) rm = fmaxf(rm, __shfl_xor(rm, d, 16));
        float mnew = fmaxf(m_i[i], rm);
        float alpha = __expf(m_i[i] - mnew);
        float p0 = __expf(sc0 - mnew), p1 = __expf(sc1 - mnew);
        float rsum = p0 + p1;
#pragma unroll
        for (int d = 1; d < 16; d <<= 1) rsum += __shfl_xor(rsum, d, 16);
        l_i[i] = l_i[i] * alpha + rsum;
        m_i[i] = mnew;
        O0[i] *= alpha; O1[i] *= alpha; O2[i] *= alpha; O3[i] *= alpha;
        pstage[wave][(i + 8 * hi) * 32 + nn]      = (bf16)p0;
        pstage[wave][(i + 8 * hi) * 32 + 16 + nn] = (bf16)p1;
      }
      wait_ds0();                         // p-tile visible to whole wave
      v16bf ap = load_a_frag(&pstage[wave][0], 32, 0);
      v16bf bv;
      bv = load_b_frag(Vt, DKn, 0, 0);  O0 = wmma_bf16(ap, bv, O0);
      bv = load_b_frag(Vt, DKn, 0, 16); O1 = wmma_bf16(ap, bv, O1);
      bv = load_b_frag(Vt, DKn, 0, 32); O2 = wmma_bf16(ap, bv, O2);
      bv = load_b_frag(Vt, DKn, 0, 48); O3 = wmma_bf16(ap, bv, O3);
      wait_ds0();
    }
  }
  float* ob = out + ((size_t)b * Ln + qbase) * DVn;
#pragma unroll
  for (int i = 0; i < 8; ++i) {
    float inv = 1.f / l_i[i];
    int M = i + 8 * hi;
    ob[(size_t)M * DVn + nn]      = O0[i] * inv;
    ob[(size_t)M * DVn + 16 + nn] = O1[i] * inv;
    ob[(size_t)M * DVn + 32 + nn] = O2[i] * inv;
    ob[(size_t)M * DVn + 48 + nn] = O3[i] * inv;
  }
}

// ---------------- host-side launch ----------------
extern "C" void kernel_launch(void* const* d_in, const int* in_sizes, int n_in,
                              void* d_out, int out_size, void* d_ws, size_t ws_size,
                              hipStream_t stream) {
  const float* x    = (const float*)d_in[0];
  const float* g_in = (const float*)d_in[10];
  const float* b_in = (const float*)d_in[11];
  const float* g_ss = (const float*)d_in[12];
  const float* b_ss = (const float*)d_in[13];
  const float* g_se = (const float*)d_in[14];
  const float* b_se = (const float*)d_in[15];
  const float* cope = (const float*)d_in[16];
  const int*   offs = (const int*)d_in[17];
  float* out = (float*)d_out;

  // workspace carve-up (256B aligned)
  uint8_t* w = (uint8_t*)d_ws;
  size_t off = 0;
  auto carve = [&](size_t bytes) { void* p = w + off; off = (off + bytes + 255) & ~(size_t)255; return p; };
  bf16* xn   = (bf16*)carve((size_t)Bn * Ln * DIn * 2);
  bf16* Wcat = (bf16*)carve((size_t)9 * DIn * DKn * 2);   // swizzled: [seg 0=in,1=ss,2=se][q,k,v]
  bf16* embb = (bf16*)carve((size_t)DKn * SLn * 2);       // swizzled
  bf16* Q    = (bf16*)carve((size_t)Bn * Ln * DKn * 2);   // pre-RoPE
  bf16* K    = (bf16*)carve((size_t)Bn * Ln * DKn * 2);
  bf16* V    = (bf16*)carve((size_t)Bn * Ln * DKn * 2);
  bf16* Qr   = (bf16*)carve((size_t)Bn * Ln * DKn * 2);
  bf16* Kr   = (bf16*)carve((size_t)Bn * Ln * DKn * 2);
  float* Tg  = (float*)carve((size_t)Bn * Sn * 4);
  float* Li  = (float*)carve((size_t)Bn * Sn * SLn * 4);

  // 1) convert + swizzle weights and cope_emb to bf16 fragment-major layout
  {
    const int nw = DIn * DKn;
    for (int j = 0; j < 9; ++j) {
      const float* src = (const float*)d_in[1 + j];
      cvt_swz_kernel<<<(nw + 255) / 256, 256, 0, stream>>>(src, Wcat + (size_t)j * nw, DIn, DKn);
    }
    const int ne = DKn * SLn;
    cvt_swz_kernel<<<(ne + 255) / 256, 256, 0, stream>>>(cope, embb, DKn, SLn);
  }
  // 2) LayerNorm
  ln_kernel<<<Bn * Ln, 256, 0, stream>>>(x, g_in, b_in, g_ss, b_ss, g_se, b_se, xn);
  // 3) QKV projections (WMMA, swizzled weights)
  qkv_kernel<<<dim3(Bn * Ln / 128, 3), 256, 0, stream>>>(xn, Wcat, Q, K, V);
  // 4) gate row totals over middle block (WMMA)
  gate_total_kernel<<<dim3(Bn, Sn / 16 / 8), 256, 0, stream>>>(Q, K, Tg);
  // 5) li = q_mid @ cope_emb (WMMA)
  li_kernel<<<dim3(Bn, Sn / 16 / 8), 256, 0, stream>>>(Q, embb, Li);
  // 6) RoPE
  rope_kernel<<<(Bn * Ln * 32 + 255) / 256, 256, 0, stream>>>(Q, K, offs, Qr, Kr);
  // 7) fused flash attention with CoPE bias (WMMA + async LDS staging)
  attn_kernel<<<dim3(Bn, Ln / 16 / 8), 256, 0, stream>>>(Qr, Kr, V, Q, K, Tg, Li, out);
  (void)in_sizes; (void)n_in; (void)out_size; (void)ws_size;
}
